// RecurrentGraphTransformer_80659485819149
// MI455X (gfx1250) — compile-verified, matrix operand order checked by reading, not run on hardware
//
#include <hip/hip_runtime.h>
#include <hip/hip_bf16.h>

// RecurrentGraphTransformer layer for MI455X (gfx1250, wave32, WMMA bf16).
// N=50000 nodes, E=800000 edges, D=128, H=8, DH=16.

#define N_NODES 50000
#define N_EDGES 800000
#define DIM 128

typedef __attribute__((ext_vector_type(16))) __bf16 v16bf;
typedef __attribute__((ext_vector_type(8)))  float  v8f;

using bf16 = __hip_bfloat16;

// ---------------------------------------------------------------- utilities

__global__ void zero_f32_kernel(float* __restrict__ p, long long n) {
  long long i = (long long)blockIdx.x * blockDim.x + threadIdx.x;
  if (i < n) p[i] = 0.0f;
}

__global__ void cvt_f32_bf16_kernel(const float* __restrict__ s,
                                    bf16* __restrict__ d, long long n) {
  long long i = (long long)blockIdx.x * blockDim.x + threadIdx.x;
  if (i < n) d[i] = __float2bfloat16(s[i]);
}

// W[K][Nc] f32 (row-major) -> Wt[Nc][K] bf16 (row-major).  Weights are tiny.
__global__ void transpose_cvt_kernel(const float* __restrict__ W,
                                     bf16* __restrict__ Wt, int K, int Nc) {
  int t = blockIdx.x * blockDim.x + threadIdx.x;
  if (t >= K * Nc) return;
  int n = t / K;
  int k = t - n * K;
  Wt[(size_t)n * K + k] = __float2bfloat16(W[(size_t)k * Nc + n]);
}

// ---------------------------------------------------------------- WMMA GEMM
// Block = 16 rows x 128 cols of output, 8 waves (256 thr), wave w owns cols
// [16w,16w+16).  A[M][K] bf16 row-major, Wt[Ncols][K] bf16 row-major (i.e. B
// column-major, so the B fragment load is identical to the A fragment load).
//
// 16-bit A-matrix 16x32 layout (ISA 7.12.2): lane L (half = L>>4) holds row
// M=L&15; VGPR pairs cover K = {khalf*8 + 0..7} U {16 + khalf*8 + 0..7}
// -> exactly two contiguous 16B loads per lane per K-step of 32.

enum { EP_PLAIN = 0, EP_BIAS_RELU = 1, EP_RES_LN = 2 };

template <int EP>
__global__ __launch_bounds__(256)
void gemm_bf16_wmma(const bf16* __restrict__ A, const bf16* __restrict__ Wt,
                    const float* __restrict__ bias, const float* __restrict__ res,
                    const float* __restrict__ gam, const float* __restrict__ bet,
                    float* __restrict__ outF, bf16* __restrict__ outB,
                    int K, int Ntot) {
  const int tid   = threadIdx.x;
  const int wave  = tid >> 5;
  const int lane  = tid & 31;
  const int khalf = lane >> 4;
  const int lrow  = lane & 15;
  const int m0    = blockIdx.x * 16;
  const int colLocal = wave * 16 + lrow;
  const int col      = blockIdx.y * 128 + colLocal;

  const bf16* Ap = A  + (size_t)(m0 + lrow) * K + khalf * 8;
  const bf16* Bp = Wt + (size_t)col * K + khalf * 8;

  v8f acc = {};
  for (int k0 = 0; k0 < K; k0 += 32) {
    v16bf a, b;
    ((uint4*)&a)[0] = *(const uint4*)(Ap + k0);
    ((uint4*)&a)[1] = *(const uint4*)(Ap + k0 + 16);
    ((uint4*)&b)[0] = *(const uint4*)(Bp + k0);
    ((uint4*)&b)[1] = *(const uint4*)(Bp + k0 + 16);
    acc = __builtin_amdgcn_wmma_f32_16x16x32_bf16(
        false, a, false, b, (short)0, acc, false, false);
  }

  // D layout: acc[i] sits at row m0 + i + khalf*8, column `col`.
  if (EP == EP_PLAIN) {
#pragma unroll
    for (int i = 0; i < 8; ++i) {
      int r = m0 + i + khalf * 8;
      outF[(size_t)r * Ntot + col] = acc[i];
    }
  } else if (EP == EP_BIAS_RELU) {
    float bc = bias[col];
#pragma unroll
    for (int i = 0; i < 8; ++i) {
      int r = m0 + i + khalf * 8;
      float v = acc[i] + bc;
      v = v > 0.0f ? v : 0.0f;
      outB[(size_t)r * Ntot + col] = __float2bfloat16(v);
    }
  } else {  // EP_RES_LN : y = res + acc + bias; LayerNorm over the 128 cols.
    __shared__ float tile[16][132];
    __shared__ float mu[16], rs[16];
    float bc = bias[col];
#pragma unroll
    for (int i = 0; i < 8; ++i) {
      int ri = i + khalf * 8;
      tile[ri][colLocal] = acc[i] + bc + res[(size_t)(m0 + ri) * 128 + col];
    }
    __syncthreads();
    if (tid < 16) {
      float s = 0.0f;
      for (int c = 0; c < 128; ++c) s += tile[tid][c];
      float m = s * (1.0f / 128.0f);
      float v = 0.0f;
      for (int c = 0; c < 128; ++c) {
        float d = tile[tid][c] - m;
        v += d * d;
      }
      v *= (1.0f / 128.0f);
      mu[tid] = m;
      rs[tid] = rsqrtf(v + 1e-5f);
    }
    __syncthreads();
#pragma unroll
    for (int t = 0; t < 8; ++t) {
      int idx = tid * 8 + t;           // 2048 elements / 256 threads
      int r = idx >> 7;
      int c = idx & 127;
      float y = (tile[r][c] - mu[r]) * rs[r] * gam[c] + bet[c];
      outF[(size_t)(m0 + r) * 128 + c] = y;
      if (outB) outB[(size_t)(m0 + r) * 128 + c] = __float2bfloat16(y);
    }
  }
}

// ------------------------------------------------------------- edge kernel
// One wave32 per edge. Lane l owns channels 4l..4l+3 (head = l>>2).
// score_h = (K[src].Q[dst])_h / sqrt(16);  e_attn = score_h * pe (bf16 out);
// sc_h = exp(clip(score_h * sum_h(pe), -5, 5));
// scatter:  wV[dst] += V[src]*sc_h ;  z[dst,h] += sc_h  (f32 atomics).

__global__ __launch_bounds__(256)
void edge_attn_kernel(const float* __restrict__ QKV, const float* __restrict__ pe,
                      const int* __restrict__ src, const int* __restrict__ dst,
                      bf16* __restrict__ eattn, float* __restrict__ wV,
                      float* __restrict__ z, int E) {
  int eid  = (int)((blockIdx.x * (long long)blockDim.x + threadIdx.x) >> 5);
  int lane = threadIdx.x & 31;
  if (eid >= E) return;
  int s = src[eid];
  int d = dst[eid];
  const float4 q = *(const float4*)(QKV + (size_t)d * 384 + 4 * lane);
  const float4 k = *(const float4*)(QKV + (size_t)s * 384 + 128 + 4 * lane);
  const float4 v = *(const float4*)(QKV + (size_t)s * 384 + 256 + 4 * lane);
  const float4 p = *(const float4*)(pe + (size_t)eid * 128 + 4 * lane);

  float dp = q.x * k.x + q.y * k.y + q.z * k.z + q.w * k.w;
  float ps = p.x + p.y + p.z + p.w;
  dp += __shfl_xor(dp, 1, 32);
  dp += __shfl_xor(dp, 2, 32);
  ps += __shfl_xor(ps, 1, 32);
  ps += __shfl_xor(ps, 2, 32);

  float score = dp * 0.25f;  // / sqrt(DH=16)

  bf16* ea = eattn + (size_t)eid * 128 + 4 * lane;
  ea[0] = __float2bfloat16(score * p.x);
  ea[1] = __float2bfloat16(score * p.y);
  ea[2] = __float2bfloat16(score * p.z);
  ea[3] = __float2bfloat16(score * p.w);

  float sh = score * ps;
  sh = fminf(fmaxf(sh, -5.0f), 5.0f);
  float sc = __expf(sh);

  float* w = wV + (size_t)d * 128 + 4 * lane;
  __hip_atomic_fetch_add(w + 0, v.x * sc, __ATOMIC_RELAXED, __HIP_MEMORY_SCOPE_AGENT);
  __hip_atomic_fetch_add(w + 1, v.y * sc, __ATOMIC_RELAXED, __HIP_MEMORY_SCOPE_AGENT);
  __hip_atomic_fetch_add(w + 2, v.z * sc, __ATOMIC_RELAXED, __HIP_MEMORY_SCOPE_AGENT);
  __hip_atomic_fetch_add(w + 3, v.w * sc, __ATOMIC_RELAXED, __HIP_MEMORY_SCOPE_AGENT);
  if ((lane & 3) == 0)
    __hip_atomic_fetch_add(z + (size_t)d * 8 + (lane >> 2), sc,
                           __ATOMIC_RELAXED, __HIP_MEMORY_SCOPE_AGENT);
}

// h_attn = wV / (z + 1e-6)   (bf16 out for the Oh GEMM)
__global__ void node_fin_kernel(const float* __restrict__ wV,
                                const float* __restrict__ z,
                                bf16* __restrict__ hattn, long long n) {
  long long i = (long long)blockIdx.x * blockDim.x + threadIdx.x;
  if (i >= n) return;
  int node = (int)(i >> 7);
  int c = (int)(i & 127);
  int h = c >> 4;
  hattn[i] = __float2bfloat16(wV[i] / (z[(size_t)node * 8 + h] + 1e-6f));
}

// ---------------------------------------------------------------- launcher

extern "C" void kernel_launch(void* const* d_in, const int* in_sizes, int n_in,
                              void* d_out, int out_size, void* d_ws, size_t ws_size,
                              hipStream_t stream) {
  const float* h    = (const float*)d_in[0];
  const float* e    = (const float*)d_in[1];
  const int*   src  = (const int*)d_in[2];
  const int*   dst  = (const int*)d_in[3];
  const float* Wq   = (const float*)d_in[4];
  const float* Wk   = (const float*)d_in[5];
  const float* Wv   = (const float*)d_in[6];
  const float* Wpe  = (const float*)d_in[7];
  const float* OhW  = (const float*)d_in[8];
  const float* OhB  = (const float*)d_in[9];
  const float* OeW  = (const float*)d_in[10];
  const float* OeB  = (const float*)d_in[11];
  const float* g1h  = (const float*)d_in[12];
  const float* b1h  = (const float*)d_in[13];
  const float* g1e  = (const float*)d_in[14];
  const float* b1e  = (const float*)d_in[15];
  const float* F1hW = (const float*)d_in[16];
  const float* F1hB = (const float*)d_in[17];
  const float* F2hW = (const float*)d_in[18];
  const float* F2hB = (const float*)d_in[19];
  const float* F1eW = (const float*)d_in[20];
  const float* F1eB = (const float*)d_in[21];
  const float* F2eW = (const float*)d_in[22];
  const float* F2eB = (const float*)d_in[23];
  const float* g2h  = (const float*)d_in[24];
  const float* b2h  = (const float*)d_in[25];
  const float* g2e  = (const float*)d_in[26];
  const float* b2e  = (const float*)d_in[27];
  float* out = (float*)d_out;

  const long long N = N_NODES, E = N_EDGES;

  char* ws = (char*)d_ws;
  size_t off = 0;
  auto alloc = [&](size_t bytes) -> void* {
    void* p = ws + off;
    off = (off + bytes + 255) & ~(size_t)255;
    return p;
  };

  bf16* hbf    = (bf16*)alloc((size_t)N * 128 * 2);
  bf16* ebf    = (bf16*)alloc((size_t)E * 128 * 2);   // reused as e_attn (bf16)
  bf16* WtQKV  = (bf16*)alloc((size_t)384 * 128 * 2);
  bf16* WtPe   = (bf16*)alloc((size_t)128 * 128 * 2);
  bf16* WtOh   = (bf16*)alloc((size_t)128 * 128 * 2);
  bf16* WtOe   = (bf16*)alloc((size_t)128 * 128 * 2);
  bf16* WtF1h  = (bf16*)alloc((size_t)256 * 128 * 2);
  bf16* WtF2h  = (bf16*)alloc((size_t)128 * 256 * 2);
  bf16* WtF1e  = (bf16*)alloc((size_t)256 * 128 * 2);
  bf16* WtF2e  = (bf16*)alloc((size_t)128 * 256 * 2);
  float* QKV   = (float*)alloc((size_t)N * 384 * 4);
  float* peb   = (float*)alloc((size_t)E * 128 * 4);  // reused as ee_f32
  float* wV    = (float*)alloc((size_t)N * 128 * 4);
  float* zb    = (float*)alloc((size_t)N * 8 * 4);
  bf16* hattn  = (bf16*)alloc((size_t)N * 128 * 2);
  float* hhF   = (float*)alloc((size_t)N * 128 * 4);
  bf16* hhB    = (bf16*)alloc((size_t)N * 128 * 2);
  bf16* hhid   = (bf16*)alloc((size_t)N * 256 * 2);
  bf16* eeB    = (bf16*)alloc((size_t)E * 128 * 2);
  bf16* ehid   = (bf16*)alloc((size_t)E * 256 * 2);
  (void)ws_size; (void)in_sizes; (void)n_in; (void)out_size;

  const int T = 256;
  auto blks = [](long long n, int t) { return (unsigned)((n + t - 1) / t); };

  // 0) zero accumulators
  zero_f32_kernel<<<blks(N * 128, T), T, 0, stream>>>(wV, N * 128);
  zero_f32_kernel<<<blks(N * 8, T), T, 0, stream>>>(zb, N * 8);

  // 1) activations + weights -> bf16 (weights transposed)
  cvt_f32_bf16_kernel<<<blks(N * 128, T), T, 0, stream>>>(h, hbf, N * 128);
  cvt_f32_bf16_kernel<<<blks(E * 128, T), T, 0, stream>>>(e, ebf, E * 128);
  transpose_cvt_kernel<<<blks(128 * 128, T), T, 0, stream>>>(Wq,  WtQKV,                 128, 128);
  transpose_cvt_kernel<<<blks(128 * 128, T), T, 0, stream>>>(Wk,  WtQKV + 128 * 128,     128, 128);
  transpose_cvt_kernel<<<blks(128 * 128, T), T, 0, stream>>>(Wv,  WtQKV + 2 * 128 * 128, 128, 128);
  transpose_cvt_kernel<<<blks(128 * 128, T), T, 0, stream>>>(Wpe, WtPe, 128, 128);
  transpose_cvt_kernel<<<blks(128 * 128, T), T, 0, stream>>>(OhW, WtOh, 128, 128);
  transpose_cvt_kernel<<<blks(128 * 128, T), T, 0, stream>>>(OeW, WtOe, 128, 128);
  transpose_cvt_kernel<<<blks(128 * 256, T), T, 0, stream>>>(F1hW, WtF1h, 128, 256);
  transpose_cvt_kernel<<<blks(256 * 128, T), T, 0, stream>>>(F2hW, WtF2h, 256, 128);
  transpose_cvt_kernel<<<blks(128 * 256, T), T, 0, stream>>>(F1eW, WtF1e, 128, 256);
  transpose_cvt_kernel<<<blks(256 * 128, T), T, 0, stream>>>(F2eW, WtF2e, 256, 128);

  // 2) fused QKV projection [N,128]x[128,384] and pe = e@Wpe [E,128]x[128,128]
  gemm_bf16_wmma<EP_PLAIN><<<dim3((unsigned)(N / 16), 3), T, 0, stream>>>(
      hbf, WtQKV, nullptr, nullptr, nullptr, nullptr, QKV, nullptr, 128, 384);
  gemm_bf16_wmma<EP_PLAIN><<<dim3((unsigned)(E / 16), 1), T, 0, stream>>>(
      ebf, WtPe, nullptr, nullptr, nullptr, nullptr, peb, nullptr, 128, 128);

  // 3) edge attention: e_attn (into ebf), scatter wV / z
  edge_attn_kernel<<<(unsigned)(E / 8), T, 0, stream>>>(QKV, peb, src, dst,
                                                        ebf, wV, zb, (int)E);

  // 4) h_attn = wV / (z + 1e-6)
  node_fin_kernel<<<blks(N * 128, T), T, 0, stream>>>(wV, zb, hattn, N * 128);

  // 5) output proj + residual + LN1  (h and e)
  gemm_bf16_wmma<EP_RES_LN><<<dim3((unsigned)(N / 16), 1), T, 0, stream>>>(
      hattn, WtOh, OhB, h, g1h, b1h, hhF, hhB, 128, 128);
  gemm_bf16_wmma<EP_RES_LN><<<dim3((unsigned)(E / 16), 1), T, 0, stream>>>(
      ebf, WtOe, OeB, e, g1e, b1e, peb, eeB, 128, 128);  // peb now holds ee

  // 6) FFN + residual + LN2 for h
  gemm_bf16_wmma<EP_BIAS_RELU><<<dim3((unsigned)(N / 16), 2), T, 0, stream>>>(
      hhB, WtF1h, F1hB, nullptr, nullptr, nullptr, nullptr, hhid, 128, 256);
  gemm_bf16_wmma<EP_RES_LN><<<dim3((unsigned)(N / 16), 1), T, 0, stream>>>(
      hhid, WtF2h, F2hB, hhF, g2h, b2h, out, nullptr, 256, 128);

  // 7) FFN + residual + LN2 for e
  gemm_bf16_wmma<EP_BIAS_RELU><<<dim3((unsigned)(E / 16), 2), T, 0, stream>>>(
      eeB, WtF1e, F1eB, nullptr, nullptr, nullptr, nullptr, ehid, 128, 256);
  gemm_bf16_wmma<EP_RES_LN><<<dim3((unsigned)(E / 16), 1), T, 0, stream>>>(
      ehid, WtF2e, F2eB, peb, g2e, b2e, out + (size_t)N_NODES * 128, nullptr, 256, 128);
}
